// ODELSTM_46892452938086
// MI455X (gfx1250) — matrix operand order repaired and checked by reading, not compiled, not run on hardware
//
#include <hip/hip_runtime.h>

// ODE-LSTM for MI455X (gfx1250): bf16 WMMA 16x16x32 with f32 accumulate.
// B=256, T=128, D=256, U=1024. Weights converted once per launch to
// transposed bf16 (L2-resident, ~14MB). x optionally pre-converted to bf16
// (16MB ws) to keep f32->bf16 VALU off the sequential chain. All recurrent
// state f32 with a bf16 shadow of h for the WMMA A-operand.

#define UNITS     1024
#define INPUT_DIM 256
#define BATCH     256
#define SEQ_T     128
#define GATES4    (4 * UNITS)

typedef __attribute__((ext_vector_type(16))) __bf16 v16bf;
typedef __attribute__((ext_vector_type(8)))  __bf16 v8bf;
typedef __attribute__((ext_vector_type(8)))  float  v8f;

#define WMMA_BF16(a, b, c) \
  __builtin_amdgcn_wmma_f32_16x16x32_bf16(false, (a), false, (b), (short)0, (c), false, false)

__device__ __forceinline__ __bf16 f2bf(float f) {
  unsigned u = __builtin_bit_cast(unsigned, f);
  u += 0x7FFFu + ((u >> 16) & 1u);            // round-to-nearest-even
  unsigned short s = (unsigned short)(u >> 16);
  return __builtin_bit_cast(__bf16, s);
}

__device__ __forceinline__ float sigm(float x) {
  return 1.0f / (1.0f + __expf(-x));
}

__device__ __forceinline__ v16bf frag_cat(v8bf lo, v8bf hi) {
  return __builtin_shufflevector(lo, hi, 0, 1, 2, 3, 4, 5, 6, 7,
                                          8, 9, 10, 11, 12, 13, 14, 15);
}

// A fragment (16x32 bf16, MxK) given a pointer to the start of this lane's row.
// ISA layout: lanes 0-15 hold K{0..7,16..23}, lanes 16-31 hold K{8..15,24..31}.
__device__ __forceinline__ v16bf load_a_row(const __bf16* __restrict__ rowp,
                                            int k0, int lane) {
  int kb = k0 + ((lane & 16) ? 8 : 0);
  const __bf16* p = rowp + kb;
  v8bf lo = *(const v8bf*)(p);        // K kb..kb+7   (16B aligned)
  v8bf hi = *(const v8bf*)(p + 16);   // K kb+16..kb+23
  return frag_cat(lo, hi);
}

// A fragment built from f32 source, converting in-register (fallback path).
__device__ __forceinline__ v16bf load_a_f32(const float* __restrict__ rowp,
                                            int k0, int lane) {
  int kb = k0 + ((lane & 16) ? 8 : 0);
  const float* p = rowp + kb;
  v16bf a;
#pragma unroll
  for (int j = 0; j < 8; ++j) {
    a[j]     = f2bf(p[j]);
    a[j + 8] = f2bf(p[j + 16]);
  }
  return a;
}

// B fragment (32x16 bf16, KxN) from PRE-TRANSPOSED weights Wt[N][ldk]:
// bptr = &Wt[(nBase + lane&15)][ (lane&16)?8:0 ]; K{kb..kb+7, kb+16..kb+23} / lane.
__device__ __forceinline__ const __bf16* b_base(const __bf16* __restrict__ Wt,
                                                int nBase, int ldk, int lane) {
  return Wt + (size_t)(nBase + (lane & 15)) * ldk + ((lane & 16) ? 8 : 0);
}

__device__ __forceinline__ v16bf load_b(const __bf16* __restrict__ bptr, int k0) {
  v8bf lo = *(const v8bf*)(bptr + k0);
  v8bf hi = *(const v8bf*)(bptr + k0 + 16);
  return frag_cat(lo, hi);
}

__device__ __forceinline__ v8f vzero8() {
  v8f z = {0.f, 0.f, 0.f, 0.f, 0.f, 0.f, 0.f, 0.f};
  return z;
}

// ---------------------------------------------------------------- prep ----
// Wt[n*K + k] = bf16(W[k*N + n])  (writes coalesced along k)
__global__ void k_transpose_bf16(const float* __restrict__ W,
                                 __bf16* __restrict__ Wt, int K, int N) {
  int total = K * N;
  for (int idx = blockIdx.x * blockDim.x + threadIdx.x; idx < total;
       idx += gridDim.x * blockDim.x) {
    int n = idx / K, k = idx - n * K;
    Wt[idx] = f2bf(W[(size_t)k * N + n]);
  }
}

// Whole-tensor f32 -> bf16 (same layout), parallel, HBM-bound, off the chain.
__global__ void k_convert_bf16(const float* __restrict__ src,
                               __bf16* __restrict__ dst, int total) {
  for (int idx = blockIdx.x * blockDim.x + threadIdx.x; idx < total;
       idx += gridDim.x * blockDim.x) {
    dst[idx] = f2bf(src[idx]);
  }
}

__global__ void k_init(float* __restrict__ cell, float* __restrict__ h,
                       __bf16* __restrict__ hbf0, __bf16* __restrict__ hbf1) {
  int total = BATCH * UNITS;
  __bf16 z = __builtin_bit_cast(__bf16, (unsigned short)0);
  for (int idx = blockIdx.x * blockDim.x + threadIdx.x; idx < total;
       idx += gridDim.x * blockDim.x) {
    cell[idx] = 0.f;
    h[idx]    = 0.f;
    hbf0[idx] = z;
    hbf1[idx] = z;
  }
}

// ---------------------------------------------------- LSTM gate step ------
// One wave per (batch 16-row tile, u 16-col tile). 4 accumulators = the four
// gates for the same u range -> A fragments shared, z never materialized.
template <bool XBF>
__global__ void __launch_bounds__(256) k_lstm_step(
    const float*  __restrict__ x,      // [B,T,D] f32
    const __bf16* __restrict__ xbf,    // [B,T,D] bf16 (if XBF)
    const __bf16* __restrict__ inT,    // [4U][D]
    const __bf16* __restrict__ recT,   // [4U][U]
    const float*  __restrict__ bias,   // [4U]
    const __bf16* __restrict__ hbf,    // [B][U]
    float*        __restrict__ cell,   // [B][U]
    __bf16*       __restrict__ odein,  // [B][U]
    int t) {
  int lane    = threadIdx.x & 31;
  int wave    = blockIdx.x * (blockDim.x >> 5) + (threadIdx.x >> 5);
  int rowBase = (wave >> 6) << 4;   // 16 batch tiles
  int uBase   = (wave & 63) << 4;   // 64 u tiles
  int arow    = rowBase + (lane & 15);
  size_t xoff = ((size_t)arow * SEQ_T + t) * INPUT_DIM;

  v8f acc[4] = {vzero8(), vzero8(), vzero8(), vzero8()};

  const __bf16* bIn[4];
  const __bf16* bRec[4];
#pragma unroll
  for (int g = 0; g < 4; ++g) {
    bIn[g]  = b_base(inT,  g * UNITS + uBase, INPUT_DIM, lane);
    bRec[g] = b_base(recT, g * UNITS + uBase, UNITS,     lane);
  }

  // x_t @ W_in  (K = 256)
  for (int k0 = 0; k0 < INPUT_DIM; k0 += 32) {
    v16bf a;
    if constexpr (XBF) a = load_a_row(xbf + xoff, k0, lane);
    else               a = load_a_f32(x + xoff, k0, lane);
#pragma unroll
    for (int g = 0; g < 4; ++g) acc[g] = WMMA_BF16(a, load_b(bIn[g], k0), acc[g]);
  }
  // h @ W_rec  (K = 1024)
  const __bf16* hrow = hbf + (size_t)arow * UNITS;
  for (int k0 = 0; k0 < UNITS; k0 += 32) {
    v16bf a = load_a_row(hrow, k0, lane);
#pragma unroll
    for (int g = 0; g < 4; ++g) acc[g] = WMMA_BF16(a, load_b(bRec[g], k0), acc[g]);
  }

  int   ucol = uBase + (lane & 15);
  float bi   = bias[0 * UNITS + ucol];
  float big  = bias[1 * UNITS + ucol];
  float bfg  = bias[2 * UNITS + ucol];
  float bog  = bias[3 * UNITS + ucol];
  int   roff = (lane & 16) ? 8 : 0;
#pragma unroll
  for (int r = 0; r < 8; ++r) {
    size_t idx = (size_t)(rowBase + r + roff) * UNITS + ucol;
    float iv  = acc[0][r] + bi;
    float igv = acc[1][r] + big;
    float fgv = acc[2][r] + bfg;
    float ogv = acc[3][r] + bog;
    float c_new = cell[idx] * sigm(fgv + 3.0f) + tanhf(iv) * sigm(igv);
    cell[idx]  = c_new;
    odein[idx] = f2bf(tanhf(c_new) * sigm(ogv));
  }
}

// -------------------------------------------- h_in = odein @ ct_kernel ----
__global__ void __launch_bounds__(256) k_hin(
    const __bf16* __restrict__ odein,  // [B][U]
    const __bf16* __restrict__ ctkT,   // [U][U]
    const float*  __restrict__ ct_bias,
    float*        __restrict__ h_in) {
  int lane    = threadIdx.x & 31;
  int wave    = blockIdx.x * (blockDim.x >> 5) + (threadIdx.x >> 5);
  int rowBase = (wave >> 6) << 4;
  int uBase   = (wave & 63) << 4;
  const __bf16* arow = odein + (size_t)(rowBase + (lane & 15)) * UNITS;
  const __bf16* bp   = b_base(ctkT, uBase, UNITS, lane);

  v8f acc = vzero8();
  for (int k0 = 0; k0 < UNITS; k0 += 32) {
    acc = WMMA_BF16(load_a_row(arow, k0, lane), load_b(bp, k0), acc);
  }
  int   ucol = uBase + (lane & 15);
  float cb   = ct_bias[ucol];
  int   roff = (lane & 16) ? 8 : 0;
#pragma unroll
  for (int r = 0; r < 8; ++r) {
    h_in[(size_t)(rowBase + r + roff) * UNITS + ucol] = acc[r] + cb;
  }
}

// ------------------------------ Euler substep: GEMM + fused update --------
// Reads h (f32) + bf16 shadow `rd`, writes h and shadow `wr` (ping-pong so no
// wave reads bf16 values another wave is overwriting).
__global__ void __launch_bounds__(256) k_substep(
    const __bf16* __restrict__ ctrT,   // [U][U]
    const float*  __restrict__ h_in,   // [B][U]
    const float*  __restrict__ ct_scale,
    float*        __restrict__ h,      // [B][U] f32 state (in/out)
    const __bf16* __restrict__ rd,     // bf16 h (read)
    __bf16*       __restrict__ wr,     // bf16 h (write)
    float*        __restrict__ out,    // [B,T,U]
    int t, int last) {
  int lane    = threadIdx.x & 31;
  int wave    = blockIdx.x * (blockDim.x >> 5) + (threadIdx.x >> 5);
  int rowBase = (wave >> 6) << 4;
  int uBase   = (wave & 63) << 4;
  const __bf16* arow = rd + (size_t)(rowBase + (lane & 15)) * UNITS;
  const __bf16* bp   = b_base(ctrT, uBase, UNITS, lane);

  v8f acc = vzero8();
  for (int k0 = 0; k0 < UNITS; k0 += 32) {
    // Prefetch next K-block of the weight panel (global_prefetch_b8): hides
    // L2 latency behind the current WMMA.
    if (k0 + 32 < UNITS) __builtin_prefetch(bp + k0 + 32, 0, 3);
    acc = WMMA_BF16(load_a_row(arow, k0, lane), load_b(bp, k0), acc);
  }

  const float dt = 1.0f / 4.0f;  // ELAPSED / NUM_UNFOLDS
  int   ucol = uBase + (lane & 15);
  float sc   = ct_scale[ucol];
  int   roff = (lane & 16) ? 8 : 0;
#pragma unroll
  for (int r = 0; r < 8; ++r) {
    int    brow = rowBase + r + roff;
    size_t idx  = (size_t)brow * UNITS + ucol;
    float  hv   = h[idx];
    float  pre  = h_in[idx] + acc[r];
    float  hn   = hv + dt * (sc * tanhf(pre) - hv);  // TAU = 1
    h[idx]  = hn;
    wr[idx] = f2bf(hn);
    if (last) out[((size_t)brow * SEQ_T + t) * UNITS + ucol] = hn;
  }
}

// ---------------------------------------------------------------- host ----
extern "C" void kernel_launch(void* const* d_in, const int* in_sizes, int n_in,
                              void* d_out, int out_size, void* d_ws, size_t ws_size,
                              hipStream_t stream) {
  const float* x        = (const float*)d_in[0];  // [256,128,256]
  const float* W_in     = (const float*)d_in[1];  // [256,4096]
  const float* W_rec    = (const float*)d_in[2];  // [1024,4096]
  const float* bias     = (const float*)d_in[3];  // [4096]
  const float* ct_k     = (const float*)d_in[4];  // [1024,1024]
  const float* ct_r     = (const float*)d_in[5];  // [1024,1024]
  const float* ct_bias  = (const float*)d_in[6];  // [1024]
  const float* ct_scale = (const float*)d_in[7];  // [1024]
  float* out = (float*)d_out;                      // [256,128,1024]

  char*  ws  = (char*)d_ws;
  size_t off = 0;
  auto carve = [&](size_t bytes) {
    char* p = ws + off;
    off = (off + bytes + 255) & ~(size_t)255;
    return p;
  };
  __bf16* recT  = (__bf16*)carve((size_t)GATES4 * UNITS * 2);     // 8 MB
  __bf16* inT   = (__bf16*)carve((size_t)GATES4 * INPUT_DIM * 2); // 2 MB
  __bf16* ctkT  = (__bf16*)carve((size_t)UNITS * UNITS * 2);      // 2 MB
  __bf16* ctrT  = (__bf16*)carve((size_t)UNITS * UNITS * 2);      // 2 MB
  float*  cell  = (float*)carve((size_t)BATCH * UNITS * 4);       // 1 MB
  float*  hst   = (float*)carve((size_t)BATCH * UNITS * 4);       // 1 MB
  float*  h_in  = (float*)carve((size_t)BATCH * UNITS * 4);       // 1 MB
  __bf16* hbf0  = (__bf16*)carve((size_t)BATCH * UNITS * 2);      // 0.5 MB
  __bf16* hbf1  = (__bf16*)carve((size_t)BATCH * UNITS * 2);      // 0.5 MB
  __bf16* odein = (__bf16*)carve((size_t)BATCH * UNITS * 2);      // 0.5 MB

  // Optional: pre-converted bf16 x (keeps cvt VALU off the recurrence).
  const int    xTotal    = BATCH * SEQ_T * INPUT_DIM;      // 8.4M elements
  const size_t xbf_bytes = (size_t)xTotal * 2;             // 16 MB
  bool    use_xbf = (off + xbf_bytes) <= ws_size;
  __bf16* xbf     = use_xbf ? (__bf16*)carve(xbf_bytes) : nullptr;
  (void)n_in; (void)in_sizes; (void)out_size;

  // Weight prep: transposed bf16 copies (L2-resident afterwards).
  k_transpose_bf16<<<2048, 256, 0, stream>>>(W_in,  inT,  INPUT_DIM, GATES4);
  k_transpose_bf16<<<2048, 256, 0, stream>>>(W_rec, recT, UNITS,     GATES4);
  k_transpose_bf16<<<1024, 256, 0, stream>>>(ct_k,  ctkT, UNITS,     UNITS);
  k_transpose_bf16<<<1024, 256, 0, stream>>>(ct_r,  ctrT, UNITS,     UNITS);
  k_init<<<256, 256, 0, stream>>>(cell, hst, hbf0, hbf1);
  if (use_xbf) k_convert_bf16<<<4096, 256, 0, stream>>>(x, xbf, xTotal);

  // 1024 wave-tiles per GEMM kernel = 128 blocks x 8 wave32s.
  const dim3 G(128), Bk(256);
  for (int t = 0; t < SEQ_T; ++t) {
    if (use_xbf)
      k_lstm_step<true><<<G, Bk, 0, stream>>>(x, xbf, inT, recT, bias, hbf0,
                                              cell, odein, t);
    else
      k_lstm_step<false><<<G, Bk, 0, stream>>>(x, xbf, inT, recT, bias, hbf0,
                                               cell, odein, t);
    k_hin<<<G, Bk, 0, stream>>>(odein, ctkT, ct_bias, h_in);
    for (int s = 0; s < 4; ++s) {
      const __bf16* rdp = (s & 1) ? hbf1 : hbf0;
      __bf16*       wrp = (s & 1) ? hbf0 : hbf1;
      k_substep<<<G, Bk, 0, stream>>>(ctrT, h_in, ct_scale, hst, rdp, wrp,
                                      out, t, (s == 3) ? 1 : 0);
    }
  }
}
